// MultiLevelGlobalContext_72670846648420
// MI455X (gfx1250) — matrix operand order repaired and checked by reading, not compiled, not run on hardware
//
#include <hip/hip_runtime.h>
#include <hip/hip_bf16.h>
#include <math.h>

typedef __attribute__((ext_vector_type(16))) _Float16 v16h;
typedef __attribute__((ext_vector_type(8)))  float    v8f;

// ---------------------------------------------------------------------------
// Tile fetch (global -> registers) and stash (convert -> LDS), kept separate
// so the compiler batches all global loads with a single wait, and so the
// main loop can overlap next-tile fetch with WMMA on the current tile.
// A tile: 64(M) x 32(K).  B tile: 32(N) x 32(K) stored transposed [n][k].
// ---------------------------------------------------------------------------
template <int TRA>
__device__ __forceinline__ void fetchA(const float* __restrict__ A, int M, int K,
                                       int lda, int bm0, int k0, int tid,
                                       bool full, float (&r)[16])
{
  if (TRA == 0) {                       // A stored (M x K)
    if (full) {
      #pragma unroll
      for (int j = 0; j < 4; ++j) {
        int e4 = j * 128 + tid;
        int rr = e4 >> 3, kc = (e4 & 7) << 2;
        const float* src = A + (long)(bm0 + rr) * lda + k0 + kc;
        #pragma unroll
        for (int q = 0; q < 4; ++q) r[j * 4 + q] = src[q];
      }
    } else {
      #pragma unroll
      for (int j = 0; j < 16; ++j) {
        int e = j * 128 + tid;
        int rr = e >> 5, kk = e & 31;
        int gm = bm0 + rr, gk = k0 + kk;
        bool inb = (gm < M) & (gk < K);
        float v = A[(long)(inb ? gm : 0) * lda + (inb ? gk : 0)];
        r[j] = inb ? v : 0.f;
      }
    }
  } else {                              // A stored (K x M)
    if (full) {
      #pragma unroll
      for (int j = 0; j < 4; ++j) {
        int e4 = j * 128 + tid;
        int kk = e4 >> 4, mc = (e4 & 15) << 2;
        const float* src = A + (long)(k0 + kk) * lda + bm0 + mc;
        #pragma unroll
        for (int q = 0; q < 4; ++q) r[j * 4 + q] = src[q];
      }
    } else {
      #pragma unroll
      for (int j = 0; j < 16; ++j) {
        int e = j * 128 + tid;
        int kk = e >> 6, rr = e & 63;
        int gm = bm0 + rr, gk = k0 + kk;
        bool inb = (gm < M) & (gk < K);
        float v = A[(long)(inb ? gk : 0) * lda + (inb ? gm : 0)];
        r[j] = inb ? v : 0.f;
      }
    }
  }
}

template <int TRA>
__device__ __forceinline__ void stashA(_Float16 (*As)[40], int tid, bool full,
                                       const float (&r)[16])
{
  if (full) {
    #pragma unroll
    for (int j = 0; j < 4; ++j) {
      int e4 = j * 128 + tid;
      if (TRA == 0) {
        int rr = e4 >> 3, kc = (e4 & 7) << 2;
        #pragma unroll
        for (int q = 0; q < 4; ++q) As[rr][kc + q] = (_Float16)r[j * 4 + q];
      } else {
        int kk = e4 >> 4, mc = (e4 & 15) << 2;
        #pragma unroll
        for (int q = 0; q < 4; ++q) As[mc + q][kk] = (_Float16)r[j * 4 + q];
      }
    }
  } else {
    #pragma unroll
    for (int j = 0; j < 16; ++j) {
      int e = j * 128 + tid;
      if (TRA == 0) { int rr = e >> 5, kk = e & 31; As[rr][kk] = (_Float16)r[j]; }
      else          { int kk = e >> 6, rr = e & 63; As[rr][kk] = (_Float16)r[j]; }
    }
  }
}

template <int TRB>
__device__ __forceinline__ void fetchB(const float* __restrict__ B, int N, int K,
                                       int ldb, int bn0, int k0, int tid,
                                       bool full, float (&r)[8])
{
  if (TRB == 0) {                       // B stored (K x N)
    if (full) {
      #pragma unroll
      for (int j = 0; j < 2; ++j) {
        int e4 = j * 128 + tid;
        int kk = e4 >> 3, nc = (e4 & 7) << 2;
        const float* src = B + (long)(k0 + kk) * ldb + bn0 + nc;
        #pragma unroll
        for (int q = 0; q < 4; ++q) r[j * 4 + q] = src[q];
      }
    } else {
      #pragma unroll
      for (int j = 0; j < 8; ++j) {
        int e = j * 128 + tid;
        int kk = e >> 5, n = e & 31;
        int gn = bn0 + n, gk = k0 + kk;
        bool inb = (gn < N) & (gk < K);
        float v = B[(long)(inb ? gk : 0) * ldb + (inb ? gn : 0)];
        r[j] = inb ? v : 0.f;
      }
    }
  } else {                              // B stored (N x K)
    if (full) {
      #pragma unroll
      for (int j = 0; j < 2; ++j) {
        int e4 = j * 128 + tid;
        int n = e4 >> 3, kc = (e4 & 7) << 2;
        const float* src = B + (long)(bn0 + n) * ldb + k0 + kc;
        #pragma unroll
        for (int q = 0; q < 4; ++q) r[j * 4 + q] = src[q];
      }
    } else {
      #pragma unroll
      for (int j = 0; j < 8; ++j) {
        int e = j * 128 + tid;
        int n = e >> 5, kk = e & 31;
        int gn = bn0 + n, gk = k0 + kk;
        bool inb = (gn < N) & (gk < K);
        float v = B[(long)(inb ? gn : 0) * ldb + (inb ? gk : 0)];
        r[j] = inb ? v : 0.f;
      }
    }
  }
}

template <int TRB>
__device__ __forceinline__ void stashB(_Float16 (*Bs)[40], int tid, bool full,
                                       const float (&r)[8])
{
  if (full) {
    #pragma unroll
    for (int j = 0; j < 2; ++j) {
      int e4 = j * 128 + tid;
      if (TRB == 0) {
        int kk = e4 >> 3, nc = (e4 & 7) << 2;
        #pragma unroll
        for (int q = 0; q < 4; ++q) Bs[nc + q][kk] = (_Float16)r[j * 4 + q];
      } else {
        int n = e4 >> 3, kc = (e4 & 7) << 2;
        #pragma unroll
        for (int q = 0; q < 4; ++q) Bs[n][kc + q] = (_Float16)r[j * 4 + q];
      }
    }
  } else {
    #pragma unroll
    for (int j = 0; j < 8; ++j) {
      int e = j * 128 + tid;
      int kk, n;
      if (TRB == 0) { kk = e >> 5; n = e & 31; }
      else          { n = e >> 5; kk = e & 31; }
      Bs[n][kk] = (_Float16)r[j];
    }
  }
}

// ---------------------------------------------------------------------------
// Generic batched GEMM, f32 in / f32 out, f16 WMMA compute (f32 accumulate),
// LDS double-buffered: one barrier per K-step, global fetch of tile t+1
// overlaps the two WMMAs of tile t.
// C[b] (M x N) = A[b] (M x K) * B[b] (K x N)   [+ C_old if accumC]
// Block = 128 threads (4 waves); block tile 64(M) x 32(N); K-step 32.
// ---------------------------------------------------------------------------
template <int TRA, int TRB>
__global__ __launch_bounds__(128)
void gemm_wmma(const float* __restrict__ A, const float* __restrict__ B,
               float* __restrict__ C,
               int M, int N, int K, int lda, int ldb, int ldc,
               long sA, long sB, long sC, int accumC)
{
  __shared__ __align__(16) _Float16 As[2][64][40];  // row stride 80B
  __shared__ __align__(16) _Float16 Bs[2][32][40];  // [n][k]

  const int tid  = threadIdx.x;
  const int wave = tid >> 5;
  const int lane = tid & 31;
  const int hi   = lane >> 4;
  const int l16  = lane & 15;
  const int bm0  = blockIdx.y * 64;
  const int bn0  = blockIdx.x * 32;
  const long bz  = (long)blockIdx.z;

  A += bz * sA;  B += bz * sB;  C += bz * sC;

  const bool fullM = (bm0 + 64 <= M);
  const bool fullN = (bn0 + 32 <= N);

  v8f acc0 = {0.f,0.f,0.f,0.f,0.f,0.f,0.f,0.f};
  v8f acc1 = {0.f,0.f,0.f,0.f,0.f,0.f,0.f,0.f};
  float ra[16]; float rb[8];

  // prologue: stage tile 0
  bool fA = fullM && (32 <= K);
  bool fB = fullN && (32 <= K);
  fetchA<TRA>(A, M, K, lda, bm0, 0, tid, fA, ra);
  fetchB<TRB>(B, N, K, ldb, bn0, 0, tid, fB, rb);
  stashA<TRA>(As[0], tid, fA, ra);
  stashB<TRB>(Bs[0], tid, fB, rb);
  __syncthreads();

  const int nk = (K + 31) >> 5;
  for (int t = 0; t < nk; ++t) {
    const int  cur     = t & 1;
    const bool hasNext = (t + 1) < nk;
    if (hasNext) {                      // issue next tile's global loads
      int k0n = (t + 1) << 5;
      fA = fullM && (k0n + 32 <= K);
      fB = fullN && (k0n + 32 <= K);
      fetchA<TRA>(A, M, K, lda, bm0, k0n, tid, fA, ra);
      fetchB<TRB>(B, N, K, ldb, bn0, k0n, tid, fB, rb);
    }

    // fragment gathers per documented gfx1250 16-bit WMMA layouts
    union { uint4 u[2]; v16h v; } fa, fb0, fb1;
    const uint4* arow = (const uint4*)(&As[cur][wave * 16 + l16][0]);
    fa.u[0] = arow[hi];            // K = hi*8 .. +7
    fa.u[1] = arow[2 + hi];        // K = 16 + hi*8 .. +7
    const uint4* brow0 = (const uint4*)(&Bs[cur][l16][0]);
    fb0.u[0] = brow0[hi];          // K = hi*8 .. +7
    fb0.u[1] = brow0[2 + hi];      // K = 16 + hi*8 .. +7
    const uint4* brow1 = (const uint4*)(&Bs[cur][16 + l16][0]);
    fb1.u[0] = brow1[hi];
    fb1.u[1] = brow1[2 + hi];

    acc0 = __builtin_amdgcn_wmma_f32_16x16x32_f16(
        false, fa.v, false, fb0.v, (short)0, acc0, false, false);
    acc1 = __builtin_amdgcn_wmma_f32_16x16x32_f16(
        false, fa.v, false, fb1.v, (short)0, acc1, false, false);

    if (hasNext) {                      // convert + stash into other buffer
      stashA<TRA>(As[cur ^ 1], tid, fA, ra);
      stashB<TRB>(Bs[cur ^ 1], tid, fB, rb);
    }
    __syncthreads();
  }

  // store D: VGPR r, lane -> (M = r + 8*hi, N = lane&15)
  #pragma unroll
  for (int half = 0; half < 2; ++half) {
    int n = bn0 + half * 16 + l16;
    if (n < N) {
      v8f acc = half ? acc1 : acc0;
      #pragma unroll
      for (int r = 0; r < 8; ++r) {
        int m = bm0 + wave * 16 + hi * 8 + r;
        if (m < M) {
          long idx = (long)m * ldc + n;
          float v = acc[r];
          if (accumC) v += C[idx];
          C[idx] = v;
        }
      }
    }
  }
}

// ---------------------------------------------------------------------------
// Column stats over rows: mode 1 -> out = colsum/max(colnorm,eps)  (for f)
//                         mode 0 -> out = max(colnorm,eps)          (for sem)
// ---------------------------------------------------------------------------
__global__ void col_stats(const float* __restrict__ X, float* __restrict__ out,
                          int rows, int cols, long sX, int mode)
{
  long b = blockIdx.y;
  int p = blockIdx.x * 256 + threadIdx.x;
  if (p >= cols) return;
  const float* x = X + b * sX + p;
  float s = 0.f, n2 = 0.f;
  for (int r = 0; r < rows; ++r) { float v = x[(long)r * cols]; s += v; n2 += v * v; }
  float nrm = fmaxf(sqrtf(n2), 1e-12f);
  out[b * (long)cols + p] = mode ? (s / nrm) : nrm;
}

// softmax over hw of sim = sem/R * S * sqrt(ci); one block per (b,n) row
__global__ void attn_softmax(const float* __restrict__ sem, const float* __restrict__ S,
                             const float* __restrict__ R, float* __restrict__ attn,
                             int ni, int hw, float sqrtci)
{
  __shared__ float red[256];
  int row = blockIdx.x;                 // b*ni + n
  int b = row / ni;
  const float* x  = sem + (long)row * hw;
  const float* Sb = S + (long)b * hw;
  const float* Rb = R + (long)b * hw;
  float* o = attn + (long)row * hw;
  int tid = threadIdx.x;

  float mx = -3.4e38f;
  for (int p = tid; p < hw; p += 256)
    mx = fmaxf(mx, x[p] / Rb[p] * Sb[p] * sqrtci);
  red[tid] = mx; __syncthreads();
  for (int s2 = 128; s2 > 0; s2 >>= 1) { if (tid < s2) red[tid] = fmaxf(red[tid], red[tid + s2]); __syncthreads(); }
  mx = red[0]; __syncthreads();

  float sum = 0.f;
  for (int p = tid; p < hw; p += 256)
    sum += __expf(x[p] / Rb[p] * Sb[p] * sqrtci - mx);
  red[tid] = sum; __syncthreads();
  for (int s2 = 128; s2 > 0; s2 >>= 1) { if (tid < s2) red[tid] += red[tid + s2]; __syncthreads(); }
  float inv = 1.f / red[0];

  for (int p = tid; p < hw; p += 256)
    o[p] = __expf(x[p] / Rb[p] * Sb[p] * sqrtci - mx) * inv;
}

// softmax over a short row (ni <= 256), in place; one block per row
__global__ void row_softmax(float* __restrict__ X, int ncols)
{
  __shared__ float red[256];
  long row = blockIdx.x;
  float* x = X + row * ncols;
  int tid = threadIdx.x;
  float v = (tid < ncols) ? x[tid] : -3.4e38f;
  red[tid] = v; __syncthreads();
  for (int s = 128; s > 0; s >>= 1) { if (tid < s) red[tid] = fmaxf(red[tid], red[tid + s]); __syncthreads(); }
  float mx = red[0]; __syncthreads();
  float e = (tid < ncols) ? __expf(v - mx) : 0.f;
  red[tid] = e; __syncthreads();
  for (int s = 128; s > 0; s >>= 1) { if (tid < s) red[tid] += red[tid + s]; __syncthreads(); }
  float inv = 1.f / red[0];
  if (tid < ncols) x[tid] = e * inv;
}

// in-place L2 normalize each column over `rows`
__global__ void l2norm_cols(float* __restrict__ X, int rows, int cols, long sX)
{
  long b = blockIdx.y;
  int p = blockIdx.x * 256 + threadIdx.x;
  if (p >= cols) return;
  float* x = X + b * sX + p;
  float n2 = 0.f;
  for (int r = 0; r < rows; ++r) { float v = x[(long)r * cols]; n2 += v * v; }
  float inv = 1.f / fmaxf(sqrtf(n2), 1e-12f);
  for (int r = 0; r < rows; ++r) x[(long)r * cols] *= inv;
}

// normalized copy of a g2 slice: L[b][o][n] = G[b][o][off+n] / ||G[b][:][off+n]||
__global__ void lc_norm(const float* __restrict__ G, float* __restrict__ L,
                        int Cdim, int ni, int ldg, long sG)
{
  long b = blockIdx.y;
  int n = blockIdx.x * 256 + threadIdx.x;
  if (n >= ni) return;
  const float* g = G + b * sG + n;
  float n2 = 0.f;
  for (int o = 0; o < Cdim; ++o) { float v = g[(long)o * ldg]; n2 += v * v; }
  float inv = 1.f / fmaxf(sqrtf(n2), 1e-12f);
  float* l = L + b * (long)Cdim * ni + n;
  for (int o = 0; o < Cdim; ++o) l[(long)o * ni] = g[(long)o * ldg] * inv;
}

// small dense layer: Y = [relu](W (MxK) * X (KxN) + bias [+ resid])
__global__ void lin_small(const float* __restrict__ W, const float* __restrict__ bias,
                          const float* __restrict__ X, const float* __restrict__ Rs,
                          float* __restrict__ Y,
                          int M, int K, int N, int ldx, long sx,
                          int ldy, long sy, int ldr, long sr, int relu)
{
  long b = blockIdx.y;
  int idx = blockIdx.x * 256 + threadIdx.x;
  if (idx >= M * N) return;
  int m = idx / N, n = idx % N;
  const float* x = X + b * sx + n;
  const float* w = W + (long)m * K;
  float s = bias ? bias[m] : 0.f;
  for (int k = 0; k < K; ++k) s += w[k] * x[(long)k * ldx];
  if (Rs) s += Rs[b * sr + (long)m * ldr + n];
  if (relu) s = fmaxf(s, 0.f);
  Y[b * sy + (long)m * ldy + n] = s;
}

// ---------------------------------------------------------------------------
extern "C" void kernel_launch(void* const* d_in, const int* in_sizes, int n_in,
                              void* d_out, int out_size, void* d_ws, size_t ws_size,
                              hipStream_t stream)
{
  (void)in_sizes; (void)n_in; (void)out_size; (void)ws_size;
  const int B = 8, C = 256, NT = 640;
  const int HWs[4] = {10000, 2500, 625, 169};
  const int CIs[4] = {256, 512, 1024, 2048};
  const int NIs[4] = {256, 192, 128, 64};
  const int OFFs[4] = {0, 256, 448, 576};

  const float* feat[4];  const float* w_sem[4]; const float* w_emb[4];
  const float* w_q[4];   const float* w_out[4]; const float* w_res[4];
  for (int i = 0; i < 4; ++i) {
    feat[i]  = (const float*)d_in[i];
    w_sem[i] = (const float*)d_in[4 + i];
    w_emb[i] = (const float*)d_in[8 + i];
    w_q[i]   = (const float*)d_in[12 + i];
    w_out[i] = (const float*)d_in[16 + i];
    w_res[i] = (const float*)d_in[20 + i];
  }
  const float* gw1 = (const float*)d_in[24]; const float* gb1 = (const float*)d_in[25];
  const float* gw2 = (const float*)d_in[26]; const float* gb2 = (const float*)d_in[27];
  const float* gw3 = (const float*)d_in[28]; const float* gb3 = (const float*)d_in[29];
  const float* mw1 = (const float*)d_in[30]; const float* mb1 = (const float*)d_in[31];
  const float* mw2 = (const float*)d_in[32]; const float* mb2 = (const float*)d_in[33];
  const float* mw3 = (const float*)d_in[34]; const float* mb3 = (const float*)d_in[35];

  // workspace carving (reused across levels; stage A and stage C alias)
  char* wsp = (char*)d_ws;
  auto alloc = [&](size_t bytes) -> float* {
    char* p = wsp; wsp += (bytes + 255) & ~(size_t)255; return (float*)p;
  };
  const size_t big = (size_t)B * C * 10000 * sizeof(float);  // 81.92 MB
  float* bufA = alloc(big);                    // sem   / scores,aw
  float* bufB = alloc(big);                    // emb   / wc
  float* bufC = alloc(big);                    // attn  / q -> qn
  float* Sv   = alloc((size_t)B * 10000 * 4);
  float* Rv   = alloc((size_t)B * 10000 * 4);
  float* ctx  = alloc((size_t)B * C * NT * 4); // concatenated contexts
  float* g1   = alloc((size_t)B * C * NT * 4); // refined + residual
  float* g2   = alloc((size_t)B * C * NT * 4); // merged gcn output
  float* h1   = alloc((size_t)B * 64 * NT * 4);
  float* h2   = alloc((size_t)B * 64 * NT * 4);
  float* lcn  = alloc((size_t)B * C * 256 * 4);

  auto gemm = [&](const float* A, const float* Bm, float* Cm, int M, int N, int K,
                  int lda, int ldb, int ldc, long sA, long sB, long sC,
                  int trA, int trB, int accum) {
    dim3 grid((N + 31) / 32, (M + 63) / 64, B);
    if (!trA && !trB)
      gemm_wmma<0,0><<<grid, 128, 0, stream>>>(A, Bm, Cm, M, N, K, lda, ldb, ldc, sA, sB, sC, accum);
    else if (!trA && trB)
      gemm_wmma<0,1><<<grid, 128, 0, stream>>>(A, Bm, Cm, M, N, K, lda, ldb, ldc, sA, sB, sC, accum);
    else if (trA && !trB)
      gemm_wmma<1,0><<<grid, 128, 0, stream>>>(A, Bm, Cm, M, N, K, lda, ldb, ldc, sA, sB, sC, accum);
    else
      gemm_wmma<1,1><<<grid, 128, 0, stream>>>(A, Bm, Cm, M, N, K, lda, ldb, ldc, sA, sB, sC, accum);
  };

  // ===================== Stage A: per-level attention -> context =============
  for (int i = 0; i < 4; ++i) {
    const int hw = HWs[i], ci = CIs[i], ni = NIs[i], off = OFFs[i];
    const float* f = feat[i];
    // sem = w_sem x f   (ni x hw)
    gemm(w_sem[i], f, bufA, ni, hw, ci, ci, hw, hw, 0, (long)ci * hw, (long)ni * hw, 0, 0, 0);
    // emb = w_emb x f   (C x hw)
    gemm(w_emb[i], f, bufB, C, hw, ci, ci, hw, hw, 0, (long)ci * hw, (long)C * hw, 0, 0, 0);
    dim3 gs((hw + 255) / 256, B);
    col_stats<<<gs, 256, 0, stream>>>(f,    Sv, ci, hw, (long)ci * hw, 1);  // fn colsum
    col_stats<<<gs, 256, 0, stream>>>(bufA, Rv, ni, hw, (long)ni * hw, 0);  // sem colnorm
    attn_softmax<<<B * ni, 256, 0, stream>>>(bufA, Sv, Rv, bufC, ni, hw, sqrtf((float)ci));
    // ctx slice = emb (Cxhw) x attn^T (hw x ni)
    gemm(bufB, bufC, ctx + off, C, ni, hw, hw, hw, NT,
         (long)C * hw, (long)ni * hw, (long)C * NT, 0, 1, 0);
  }

  // ===================== Stage B: GCN refine + merged GCN ====================
  for (int i = 0; i < 4; ++i) {
    const int ni = NIs[i], off = OFFs[i];
    dim3 gA((64 * ni + 255) / 256, B);
    lin_small<<<gA, 256, 0, stream>>>(gw1 + (long)i * 64 * C, gb1 + i * 64,
        ctx + off, nullptr, h1, 64, C, ni, NT, (long)C * NT, ni, (long)64 * ni, 0, 0, 1);
    lin_small<<<gA, 256, 0, stream>>>(gw2 + (long)i * 64 * 64, gb2 + i * 64,
        h1, nullptr, h2, 64, 64, ni, ni, (long)64 * ni, ni, (long)64 * ni, 0, 0, 1);
    dim3 gB((C * ni + 255) / 256, B);
    lin_small<<<gB, 256, 0, stream>>>(gw3 + (long)i * C * 64, gb3 + i * C,
        h2, ctx + off, g1 + off, C, 64, ni, ni, (long)64 * ni,
        NT, (long)C * NT, NT, (long)C * NT, 0);
  }
  {
    dim3 gA((64 * NT + 255) / 256, B);
    lin_small<<<gA, 256, 0, stream>>>(mw1, mb1, g1, nullptr, h1,
        64, C, NT, NT, (long)C * NT, NT, (long)64 * NT, 0, 0, 1);
    lin_small<<<gA, 256, 0, stream>>>(mw2, mb2, h1, nullptr, h2,
        64, 64, NT, NT, (long)64 * NT, NT, (long)64 * NT, 0, 0, 1);
    dim3 gB((C * NT + 255) / 256, B);
    lin_small<<<gB, 256, 0, stream>>>(mw3, mb3, h2, g1, g2,
        C, 64, NT, NT, (long)64 * NT, NT, (long)C * NT, NT, (long)C * NT, 0);
  }

  // ===================== Stage C: second attention -> outputs ================
  long outOff = 0;
  for (int i = 0; i < 4; ++i) {
    const int hw = HWs[i], ci = CIs[i], ni = NIs[i], off = OFFs[i];
    const float* f = feat[i];
    float* outp = (float*)d_out + outOff;
    // q = w_q x f  -> bufC, then normalize columns in place (qn)
    gemm(w_q[i], f, bufC, C, hw, ci, ci, hw, hw, 0, (long)ci * hw, (long)C * hw, 0, 0, 0);
    dim3 gs((hw + 255) / 256, B);
    l2norm_cols<<<gs, 256, 0, stream>>>(bufC, C, hw, (long)C * hw);
    // lcn = normalized g2 slice (C x ni, compact)
    dim3 gl((ni + 255) / 256, B);
    lc_norm<<<gl, 256, 0, stream>>>(g2 + off, lcn, C, ni, NT, (long)C * NT);
    // scores (hw x ni) = qn^T x lcn     (A transposed layout)
    gemm(bufC, lcn, bufA, hw, ni, C, hw, ni, ni,
         (long)C * hw, (long)C * ni, (long)hw * ni, 1, 0, 0);
    row_softmax<<<B * hw, 256, 0, stream>>>(bufA, ni);   // aw, in place
    // wc (C x hw) = lc (C x ni) x aw^T  (B transposed layout)
    gemm(g2 + off, bufA, bufB, C, hw, ni, NT, ni, hw,
         (long)C * NT, (long)hw * ni, (long)C * hw, 0, 1, 0);
    // out = w_res x f ; out += w_out x wc
    gemm(w_res[i], f, outp, C, hw, ci, ci, hw, hw, 0, (long)ci * hw, (long)C * hw, 0, 0, 0);
    gemm(w_out[i], bufB, outp, C, hw, C, C, hw, hw, 0, (long)C * hw, (long)C * hw, 0, 0, 1);
    outOff += (long)B * C * hw;
  }
}
